// TritonSubMConv3d_45681272160692
// MI455X (gfx1250) — matrix operand order repaired and checked by reading, not compiled, not run on hardware
//
#include <hip/hip_runtime.h>
#include <hip/hip_bf16.h>

typedef __attribute__((ext_vector_type(16))) _Float16 v16h;
typedef __attribute__((ext_vector_type(8)))  _Float16 v8h;
typedef __attribute__((ext_vector_type(8)))  float    v8f;
typedef __attribute__((ext_vector_type(4)))  unsigned int u32x4;
typedef __attribute__((ext_vector_type(4)))  int      i32x4;
typedef __attribute__((ext_vector_type(8)))  int      i32x8;

#define HCAP   (1u << 18)
#define HMASK  (HCAP - 1u)
#define HEMPTY 0xFFFFFFFFFFFFFFFFull

// LDS layout for staged weights: 27*64 = 1728 rows of 64 halves (128B),
// padded by TDM to 144B stride -> conflict-free b128 reads.
#define W_ROWS       1728
#define W_ROW_BYTES  144
#define W_LDS_BYTES  (W_ROWS * W_ROW_BYTES)   // 248832 B <= 320KB WGP LDS

// ---------------- hash table build (unchanged) --------

__device__ __forceinline__ unsigned long long pack_key(long long x, long long y, long long z) {
  return (((unsigned long long)x) << 40) |
         (((unsigned long long)y) << 20) |
          ((unsigned long long)z);
}

__device__ __forceinline__ unsigned hash_key(unsigned long long k) {
  k *= 0x9E3779B97F4A7C15ull;
  k ^= k >> 32;
  k *= 0xBF58476D1CE4E5B9ull;
  k ^= k >> 29;
  return (unsigned)k & HMASK;
}

__global__ void hash_init(unsigned long long* __restrict__ hkeys, int* __restrict__ hvals) {
  unsigned i = blockIdx.x * blockDim.x + threadIdx.x;
  if (i < HCAP) { hkeys[i] = HEMPTY; hvals[i] = 0x7FFFFFFF; }
}

__global__ void hash_insert(const int* __restrict__ coords, const int* __restrict__ batch,
                            unsigned long long* __restrict__ hkeys, int* __restrict__ hvals,
                            int n) {
  int i = blockIdx.x * blockDim.x + threadIdx.x;
  if (i >= n) return;
  long long x = (long long)coords[i * 3 + 0] + (long long)batch[i] * 100000ll;
  long long y = coords[i * 3 + 1];
  long long z = coords[i * 3 + 2];
  unsigned long long key = pack_key(x, y, z);
  unsigned s = hash_key(key);
  for (;;) {
    unsigned long long old = atomicCAS(&hkeys[s], HEMPTY, key);
    if (old == HEMPTY || old == key) { atomicMin(&hvals[s], i); break; }
    s = (s + 1) & HMASK;
  }
}

__global__ void build_table(const int* __restrict__ coords, const int* __restrict__ batch,
                            const unsigned long long* __restrict__ hkeys,
                            const int* __restrict__ hvals,
                            int* __restrict__ table, int n) {
  int t = blockIdx.x * blockDim.x + threadIdx.x;
  if (t >= n * 27) return;
  int i = t / 27;
  int k = t - i * 27;
  int dx = k / 9 - 1, dy = (k / 3) % 3 - 1, dz = k % 3 - 1;
  long long x = (long long)coords[i * 3 + 0] + (long long)batch[i] * 100000ll + dx;
  long long y = (long long)coords[i * 3 + 1] + dy;
  long long z = (long long)coords[i * 3 + 2] + dz;
  unsigned long long key = pack_key(x, y, z);
  unsigned s = hash_key(key);
  int res = -1;
  for (;;) {
    unsigned long long kk = hkeys[s];
    if (kk == key) { res = hvals[s]; break; }
    if (kk == HEMPTY) break;
    s = (s + 1) & HMASK;
  }
  table[t] = res;
}

__global__ void cvt_feat(const float* __restrict__ f, _Float16* __restrict__ h, int total) {
  int i = blockIdx.x * blockDim.x + threadIdx.x;
  if (i < total) h[i] = (_Float16)f[i];
}

// weight (27, Cin, Cout) fp32 -> wt16 (27, Cout, Cin) f16: each lane's
// B fragment (fixed N, 16 consecutive K) is contiguous.
__global__ void cvt_weight(const float* __restrict__ w, _Float16* __restrict__ wt) {
  int i = blockIdx.x * blockDim.x + threadIdx.x;
  if (i >= 27 * 64 * 64) return;
  int k = i >> 12;
  int o = (i >> 6) & 63;
  int c = i & 63;
  wt[i] = (_Float16)w[(k << 12) + (c << 6) + o];
}

// ---------------- WMMA GEMM with TDM-staged weights in LDS -----------------
// 512 threads = 16 waves/block, one 16-row M-tile per wave.
// Wave 0 DMAs all 27 taps of fragment-layout weights (216KB) into LDS with a
// single 2-D tensor_load_to_lds (pad 4 dwords / 32 dwords -> 144B row stride,
// conflict-free banks), waits TENSORcnt, then the block barrier publishes it.
__global__ __launch_bounds__(512) void spconv_wmma(
    const _Float16* __restrict__ feat16,
    const _Float16* __restrict__ wt16,
    const int* __restrict__ table,
    float* __restrict__ out, int n) {
  __shared__ char ldsw[W_LDS_BYTES];

  const int wave = threadIdx.x >> 5;
  const int lane = threadIdx.x & 31;

  if (wave == 0) {
    unsigned long long ga = (unsigned long long)(uintptr_t)wt16;
    unsigned lds_base = (unsigned)(uintptr_t)&ldsw[0];   // low 32 bits = LDS offset
    // D# group0: count=1, lds_addr, global_addr[56:0], type=2 (bits 127:126)
    u32x4 g0 = { 0x1u, lds_base, (unsigned)ga,
                 (unsigned)(ga >> 32) | 0x80000000u };
    // D# group1:
    //  w0: data_size=1(2B)<<16 | pad_enable<<20 | pad_interval=4(32dw)<<22 | pad_amount=3(4dw)<<25
    //  w1: tensor_dim0(4096)[15:0]<<16   w2: tensor_dim1(27)[15:0]<<16
    //  w3: tile_dim0(4096)<<16           w4: tile_dim1=27
    //  w5: tensor_dim0_stride=4096       w6,w7: 0
    i32x8 g1 = { 0x07110000, 0x10000000, 0x001B0000, 0x10000000,
                 27, 4096, 0, 0 };
    i32x4 g2 = { 0, 0, 0, 0 };
    i32x4 g3 = { 0, 0, 0, 0 };
#if defined(__clang_major__) && (__clang_major__ >= 23)
    i32x8 g4 = { 0, 0, 0, 0, 0, 0, 0, 0 };
    __builtin_amdgcn_tensor_load_to_lds(g0, g1, g2, g3, g4, 0);
#else
    __builtin_amdgcn_tensor_load_to_lds(g0, g1, g2, g3, 0);
#endif
    __builtin_amdgcn_s_wait_tensorcnt(0);
  }
  __syncthreads();

  const int tile = blockIdx.x * 16 + wave;
  const int ntiles = (n + 15) >> 4;
  if (tile >= ntiles) return;            // wave-uniform: EXEC all-ones inside

  const int m  = lane & 15;              // A row / B,C,D column within 16
  const int hf = lane >> 4;              // lane half selects K sub-range
  const int rb = tile << 4;
  int gi = rb + m;
  if (gi > n - 1) gi = n - 1;

  v8f acc[4] = {};
  const int koff = hf * 16;

  for (int k = 0; k < 27; ++k) {
    const int row = table[gi * 27 + k];
    const _Float16* arow = feat16 + (long)row * 64;
#pragma unroll
    for (int kk = 0; kk < 2; ++kk) {
      v8h lo = {}, hi = {};
      if (row >= 0) {                    // converges before WMMA issue
        lo = *(const v8h*)(arow + kk * 32 + koff);
        hi = *(const v8h*)(arow + kk * 32 + 8 + koff);
      }
      union { struct { v8h lo, hi; } p; v16h v; } ua;
      ua.p.lo = lo; ua.p.hi = hi;
      const v16h a = ua.v;
      const char* brow = ldsw + (kk * 32 + hf * 16) * 2;
#pragma unroll
      for (int nt = 0; nt < 4; ++nt) {
        const char* bp = brow + (k * 64 + nt * 16 + m) * W_ROW_BYTES;
        union { struct { v8h lo, hi; } p; v16h v; } ub;
        ub.p.lo = *(const v8h*)bp;       // ds_load_b128
        ub.p.hi = *(const v8h*)(bp + 16);
        acc[nt] = __builtin_amdgcn_wmma_f32_16x16x32_f16(
            false, a, false, ub.v, (short)0, acc[nt], false, false);
      }
    }
  }

  float* op = out + (long)rb * 64;
#pragma unroll
  for (int nt = 0; nt < 4; ++nt) {
#pragma unroll
    for (int i = 0; i < 8; ++i) {
      int r = hf * 8 + i;                // C/D layout: M = 8*half + vgpr
      if (rb + r < n) op[r * 64 + nt * 16 + m] = acc[nt][i];
    }
  }
}

extern "C" void kernel_launch(void* const* d_in, const int* in_sizes, int n_in,
                              void* d_out, int out_size, void* d_ws, size_t ws_size,
                              hipStream_t stream) {
  const float* features = (const float*)d_in[0];
  const int*   coords   = (const int*)d_in[1];
  const int*   batch    = (const int*)d_in[2];
  const float* weight   = (const float*)d_in[3];
  float* out = (float*)d_out;
  const int n = in_sizes[0] / 64;       // C_IN = 64

  char* ws = (char*)d_ws;
  size_t off = 0;
  unsigned long long* hkeys = (unsigned long long*)(ws + off); off += (size_t)HCAP * 8;
  int* hvals = (int*)(ws + off);                               off += (size_t)HCAP * 4;
  int* table = (int*)(ws + off);                               off += (size_t)n * 27 * 4;
  off = (off + 255) & ~(size_t)255;
  _Float16* feat16 = (_Float16*)(ws + off);                    off += (size_t)n * 64 * 2;
  off = (off + 255) & ~(size_t)255;
  _Float16* wt16 = (_Float16*)(ws + off);                      off += (size_t)27 * 64 * 64 * 2;
  (void)ws_size; (void)n_in; (void)out_size;

  const int B = 256;
  hipLaunchKernelGGL(hash_init,   dim3((HCAP + B - 1) / B),        dim3(B), 0, stream, hkeys, hvals);
  hipLaunchKernelGGL(hash_insert, dim3((n + B - 1) / B),           dim3(B), 0, stream, coords, batch, hkeys, hvals, n);
  hipLaunchKernelGGL(build_table, dim3((n * 27 + B - 1) / B),      dim3(B), 0, stream, coords, batch, hkeys, hvals, table, n);
  hipLaunchKernelGGL(cvt_feat,    dim3((n * 64 + B - 1) / B),      dim3(B), 0, stream, features, feat16, n * 64);
  hipLaunchKernelGGL(cvt_weight,  dim3((27 * 64 * 64 + B - 1) / B), dim3(B), 0, stream, weight, wt16);
  const int ntiles = (n + 15) / 16;
  hipLaunchKernelGGL(spconv_wmma, dim3((ntiles + 15) / 16),        dim3(512), 0, stream, feat16, wt16, table, out, n);
}
